// AttentionalPropagation_71683004170881
// MI455X (gfx1250) — compile-verified, matrix operand order checked by reading, not run on hardware
//
#include <hip/hip_runtime.h>
#include <hip/hip_bf16.h>
#include <math.h>

typedef __attribute__((ext_vector_type(16))) __bf16 v16bf;
typedef __attribute__((ext_vector_type(8)))  float  v8f;
typedef __attribute__((ext_vector_type(4)))  float  f32x4;

// ---------------------------------------------------------------------------
// helpers: packed bf16 pair <-> f32
// ---------------------------------------------------------------------------
__device__ __forceinline__ unsigned pack2bf(float lo, float hi) {
    unsigned short a = __builtin_bit_cast(unsigned short, (__bf16)lo);
    unsigned short b = __builtin_bit_cast(unsigned short, (__bf16)hi);
    return ((unsigned)b << 16) | (unsigned)a;
}

// load 12 bf16 (24B, 8B-aligned LDS) -> 12 f32 via 3x ds_load_b64
__device__ __forceinline__ void load12bf(const __bf16* p, float* o) {
    const uint2* u = (const uint2*)p;
    uint2 a = u[0], b = u[1], c = u[2];
    unsigned w[6] = {a.x, a.y, b.x, b.y, c.x, c.y};
    #pragma unroll
    for (int i = 0; i < 6; ++i) {
        o[2 * i]     = __uint_as_float(w[i] << 16);
        o[2 * i + 1] = __uint_as_float(w[i] & 0xffff0000u);
    }
}

// ---------------------------------------------------------------------------
// Kernel 0: convert [K][N] f32 row-major weight into bf16 packed in WMMA
// B-fragment order: frag (nt,kt) = 32 lanes x 16 bf16; element j of lane l is
// W[kt*32 + (l>>4)*16 + j][nt*16 + (l&15)].  Each lane then loads its whole
// fragment as 32 contiguous bytes.
// ---------------------------------------------------------------------------
__global__ void pack_w(const float* __restrict__ w, __bf16* __restrict__ dst,
                       int K, int N) {
    const int ktiles = K >> 5;
    const int total  = (N >> 4) * ktiles * 512;
    int e = blockIdx.x * blockDim.x + threadIdx.x;
    if (e >= total) return;
    int j    = e & 15;
    int l    = (e >> 4) & 31;
    int frag = e >> 9;
    int kt   = frag % ktiles;
    int nt   = frag / ktiles;
    int k = kt * 32 + (l >> 4) * 16 + j;
    int n = nt * 16 + (l & 15);
    dst[e] = (__bf16)w[k * N + n];
}

// ---------------------------------------------------------------------------
// Fused kernel: one workgroup = 8 windows = 64 tokens (16-wide contiguous T
// slab).  8 waves of 32 lanes.  All four big GEMMs use v_wmma_f32_16x16x32_bf16.
// ---------------------------------------------------------------------------
__global__ __launch_bounds__(256)
void fused_swin3d(const float* __restrict__ x,
                  const float* __restrict__ qkv_b,
                  const float* __restrict__ merge_b,
                  const float* __restrict__ b1,
                  const float* __restrict__ b2,
                  const __bf16* __restrict__ pk_qkv,
                  const __bf16* __restrict__ pk_merge,
                  const __bf16* __restrict__ pk_w1,
                  const __bf16* __restrict__ pk_w2,
                  float* __restrict__ out) {
    // LDS layout (60 KB, aliased across phases):
    //   [0     .. 12288)  Xbuf : [64][96]  bf16 tokens    (later: Obuf f32 part)
    //   [12288 .. 49152)  Sbuf : [64][288] bf16 QKV       (later: mt [64][96])
    //   [49152 .. 61440)  Mbuf : [64][96]  bf16 msg       (later: hmid)
    __shared__ __align__(32) char smem[61440];
    __bf16* Xbuf = (__bf16*)smem;
    __bf16* Sbuf = (__bf16*)(smem + 12288);
    __bf16* Mbuf = (__bf16*)(smem + 49152);
    float*  Obuf = (float*)smem;

    const int tid  = threadIdx.x;
    const int lane = tid & 31;
    const int wave = tid >> 5;
    const int hk   = lane >> 4;     // K-half selector for A fragments
    const int lr   = lane & 15;

    // block -> (b, window coords). 8 windows consecutive along T per WG.
    const int g   = blockIdx.x;
    const int twq = g & 3;
    const int ww  = (g >> 2) & 31;
    const int hw  = (g >> 7) & 31;
    const int b   = g >> 12;
    const int h0 = hw * 2, w0 = ww * 2, t0 = twq * 16;

    // ---- Phase 0: load x tile (f32 HBM -> bf16 LDS), 128-bit NT loads ----
    for (int idx = tid; idx < 1536; idx += 256) {
        int q    = idx & 3;                 // float4 index within 16-wide T run
        int rest = idx >> 2;
        int dw   = rest & 1;
        int dh   = (rest >> 1) & 1;
        int c    = rest >> 2;               // 0..95
        const f32x4* src = (const f32x4*)&x[(((b * 96 + c) * 64 + (h0 + dh)) * 64
                                            + (w0 + dw)) * 64 + t0 + q * 4];
        f32x4 v = __builtin_nontemporal_load(src);
        #pragma unroll
        for (int e = 0; e < 4; ++e) {
            int t_off = q * 4 + e;
            int tok = (t_off >> 1) * 8 + dh * 4 + dw * 2 + (t_off & 1);
            Xbuf[tok * 96 + c] = (__bf16)v[e];
        }
    }
    __syncthreads();

    const float scale = 0.28867513459481287f;  // hd^-0.5, hd=12

    // ---- Phase 1: QKV = X[64x96] @ Wqkv[96x288] + b  (72 tiles, K=3) ----
    for (int tile = wave; tile < 72; tile += 8) {
        int mt = tile / 18, nt = tile % 18;
        v8f acc = {};
        for (int kt = 0; kt < 3; ++kt) {
            v16bf a  = *(const v16bf*)&Xbuf[(mt * 16 + lr) * 96 + kt * 32 + hk * 16];
            v16bf bm = *(const v16bf*)&pk_qkv[(nt * 3 + kt) * 512 + lane * 16];
            acc = __builtin_amdgcn_wmma_f32_16x16x32_bf16(false, a, false, bm,
                                                          (short)0, acc, false, false);
        }
        int col   = nt * 16 + lr;
        float bia = qkv_b[col];
        float sc  = (col < 96) ? scale : 1.0f;  // fold q-scale into Q
        #pragma unroll
        for (int v = 0; v < 8; ++v) {
            int row = mt * 16 + hk * 8 + v;
            Sbuf[row * 288 + col] = (__bf16)((acc[v] + bia) * sc);
        }
    }
    __syncthreads();

    // ---- Phase 2: windowed softmax attention (8x8 per head, VALU f32) ----
    {
        int win  = tid >> 5;        // 0..7 local window
        int rem  = tid & 31;
        int head = rem >> 2;        // 0..7
        int q2   = rem & 3;
        for (int qq = 0; qq < 2; ++qq) {
            int qt  = q2 + qq * 4;
            int row = win * 8 + qt;
            float qv[12];
            load12bf(&Sbuf[row * 288 + head * 12], qv);
            float s[8];
            float mx = -1e30f;
            #pragma unroll
            for (int j = 0; j < 8; ++j) {
                float kv[12];
                load12bf(&Sbuf[(win * 8 + j) * 288 + 96 + head * 12], kv);
                float a = 0.f;
                #pragma unroll
                for (int d = 0; d < 12; ++d) a += qv[d] * kv[d];
                s[j] = a;
                mx = fmaxf(mx, a);
            }
            float sum = 0.f;
            #pragma unroll
            for (int j = 0; j < 8; ++j) { s[j] = __expf(s[j] - mx); sum += s[j]; }
            float inv = 1.0f / sum;
            float o[12] = {0.f};
            #pragma unroll
            for (int j = 0; j < 8; ++j) {
                float p = s[j] * inv;
                float vv[12];
                load12bf(&Sbuf[(win * 8 + j) * 288 + 192 + head * 12], vv);
                #pragma unroll
                for (int d = 0; d < 12; ++d) o[d] += p * vv[d];
            }
            // pack 12 bf16 -> 3x ds_store_b64
            uint2* dst = (uint2*)&Mbuf[row * 96 + head * 12];
            uint2 d0, d1, d2;
            d0.x = pack2bf(o[0], o[1]);  d0.y = pack2bf(o[2], o[3]);
            d1.x = pack2bf(o[4], o[5]);  d1.y = pack2bf(o[6], o[7]);
            d2.x = pack2bf(o[8], o[9]);  d2.y = pack2bf(o[10], o[11]);
            dst[0] = d0; dst[1] = d1; dst[2] = d2;
        }
    }
    __syncthreads();

    // ---- Phase 3: mt = msg[64x96] @ merge_w[96x96] + b  (24 tiles) ----
    __bf16* Mt = Sbuf;   // reuse QKV region as [64][96]
    for (int tile = wave; tile < 24; tile += 8) {
        int mti = tile / 6, nt = tile % 6;
        v8f acc = {};
        for (int kt = 0; kt < 3; ++kt) {
            v16bf a  = *(const v16bf*)&Mbuf[(mti * 16 + lr) * 96 + kt * 32 + hk * 16];
            v16bf bm = *(const v16bf*)&pk_merge[(nt * 3 + kt) * 512 + lane * 16];
            acc = __builtin_amdgcn_wmma_f32_16x16x32_bf16(false, a, false, bm,
                                                          (short)0, acc, false, false);
        }
        int col   = nt * 16 + lr;
        float bia = merge_b[col];
        #pragma unroll
        for (int v = 0; v < 8; ++v) {
            int row = mti * 16 + hk * 8 + v;
            Mt[row * 96 + col] = (__bf16)(acc[v] + bia);
        }
    }
    __syncthreads();

    // ---- Phase 4: hmid = GELU(concat([X, mt])[64x192] @ w1[192x96] + b1) ----
    for (int tile = wave; tile < 24; tile += 8) {
        int mti = tile / 6, nt = tile % 6;
        v8f acc = {};
        for (int kt = 0; kt < 6; ++kt) {
            const __bf16* ap = (kt < 3)
                ? &Xbuf[(mti * 16 + lr) * 96 + kt * 32 + hk * 16]
                : &Mt  [(mti * 16 + lr) * 96 + (kt - 3) * 32 + hk * 16];
            v16bf a  = *(const v16bf*)ap;
            v16bf bm = *(const v16bf*)&pk_w1[(nt * 6 + kt) * 512 + lane * 16];
            acc = __builtin_amdgcn_wmma_f32_16x16x32_bf16(false, a, false, bm,
                                                          (short)0, acc, false, false);
        }
        int col   = nt * 16 + lr;
        float bia = b1[col];
        #pragma unroll
        for (int v = 0; v < 8; ++v) {
            int row = mti * 16 + hk * 8 + v;
            float z = acc[v] + bia;
            float gl = 0.5f * z * (1.0f + erff(z * 0.70710678118654752f));
            Mbuf[row * 96 + col] = (__bf16)gl;   // hmid overwrites msg
        }
    }
    __syncthreads();

    // ---- Phase 5: out = hmid[64x96] @ w2[96x96] + b2 -> f32 staging ----
    for (int tile = wave; tile < 24; tile += 8) {
        int mti = tile / 6, nt = tile % 6;
        v8f acc = {};
        for (int kt = 0; kt < 3; ++kt) {
            v16bf a  = *(const v16bf*)&Mbuf[(mti * 16 + lr) * 96 + kt * 32 + hk * 16];
            v16bf bm = *(const v16bf*)&pk_w2[(nt * 3 + kt) * 512 + lane * 16];
            acc = __builtin_amdgcn_wmma_f32_16x16x32_bf16(false, a, false, bm,
                                                          (short)0, acc, false, false);
        }
        int col   = nt * 16 + lr;
        float bia = b2[col];
        #pragma unroll
        for (int v = 0; v < 8; ++v) {
            int row = mti * 16 + hk * 8 + v;
            Obuf[row * 96 + col] = acc[v] + bia;   // aliases dead Xbuf/Mt region
        }
    }
    __syncthreads();

    // ---- Phase 6: coalesced 128-bit NT scatter to out [B,C,H,W,T] ----
    for (int idx = tid; idx < 1536; idx += 256) {
        int q    = idx & 3;
        int rest = idx >> 2;
        int dw   = rest & 1;
        int dh   = (rest >> 1) & 1;
        int c    = rest >> 2;
        f32x4 v;
        #pragma unroll
        for (int e = 0; e < 4; ++e) {
            int t_off = q * 4 + e;
            int tok = (t_off >> 1) * 8 + dh * 4 + dw * 2 + (t_off & 1);
            v[e] = Obuf[tok * 96 + c];
        }
        f32x4* dst = (f32x4*)&out[(((b * 96 + c) * 64 + (h0 + dh)) * 64
                                   + (w0 + dw)) * 64 + t0 + q * 4];
        __builtin_nontemporal_store(v, dst);
    }
}

// ---------------------------------------------------------------------------
extern "C" void kernel_launch(void* const* d_in, const int* in_sizes, int n_in,
                              void* d_out, int out_size, void* d_ws, size_t ws_size,
                              hipStream_t stream) {
    (void)in_sizes; (void)n_in; (void)out_size; (void)ws_size;
    const float* x       = (const float*)d_in[0];
    const float* qkv_w   = (const float*)d_in[1];
    const float* qkv_b   = (const float*)d_in[2];
    const float* merge_w = (const float*)d_in[3];
    const float* merge_b = (const float*)d_in[4];
    const float* w1      = (const float*)d_in[5];
    const float* b1      = (const float*)d_in[6];
    const float* w2      = (const float*)d_in[7];
    const float* b2      = (const float*)d_in[8];
    float* out = (float*)d_out;

    // workspace layout (bf16 elems): qkv 27648 | merge 9216 | w1 18432 | w2 9216
    char* ws = (char*)d_ws;
    __bf16* pk_qkv   = (__bf16*)(ws);
    __bf16* pk_merge = (__bf16*)(ws + 27648 * 2);
    __bf16* pk_w1    = (__bf16*)(ws + 36864 * 2);
    __bf16* pk_w2    = (__bf16*)(ws + 55296 * 2);

    pack_w<<<108, 256, 0, stream>>>(qkv_w,   pk_qkv,    96, 288);
    pack_w<<< 36, 256, 0, stream>>>(merge_w, pk_merge,  96,  96);
    pack_w<<< 72, 256, 0, stream>>>(w1,      pk_w1,    192,  96);
    pack_w<<< 36, 256, 0, stream>>>(w2,      pk_w2,     96,  96);

    fused_swin3d<<<8192, 256, 0, stream>>>(x, qkv_b, merge_b, b1, b2,
                                           pk_qkv, pk_merge, pk_w1, pk_w2, out);
}
